// EPMoELayer_52347061404316
// MI455X (gfx1250) — compile-verified, matrix operand order checked by reading, not run on hardware
//
#include <hip/hip_runtime.h>
#include <hip/hip_bf16.h>

// ---------------------------------------------------------------------------
// MoE FFN (top-2 of 8 experts + shared expert) for MI455X (gfx1250, wave32).
// Routed bf16 WMMA (v_wmma_f32_16x16x32_bf16, f32 accumulate).
//  - A operands (x_bf16 / h_bf16, L2-resident) loaded straight from global in
//    WMMA fragment layout (two per-lane global_load_b128 per 16x32 fragment),
//    which also implements the per-token gather.
//  - Only B (fp32 weights -> bf16, transposed) is staged through LDS; packed
//    v2bf stores keep staging at b32 granularity.
//  - 64x64 wave tiles: 16 WMMAs per 8 LDS fragment loads.
// ---------------------------------------------------------------------------

typedef __attribute__((ext_vector_type(16))) __bf16 v16bf;
typedef __attribute__((ext_vector_type(8)))  __bf16 v8bf;
typedef __attribute__((ext_vector_type(2)))  __bf16 v2bf;
typedef __attribute__((ext_vector_type(8)))  float  v8f;

#define N_TOK 8192
#define C_DIM 2048
#define F_DIM 4096
#define E_NUM 8

#define BM 256
#define BN 128
#define BK 32
#define LDSS 40   // LDS row stride (bf16 elems): 80B rows, 16B-aligned, conflict-free

// ---------------------------------------------------------------------------
__global__ void zero_counts_kernel(int* __restrict__ counts) {
    if (threadIdx.x < E_NUM) counts[threadIdx.x] = 0;
}

// ---------------------------------------------------------------------------
// Gate: per token -> 8 logits, top-2, renormalized weights; also x -> bf16.
__global__ __launch_bounds__(256) void gate_kernel(
    const float* __restrict__ x, const float* __restrict__ Wg,
    __bf16* __restrict__ xb, int* __restrict__ counts,
    int* __restrict__ tokList, float* __restrict__ wList)
{
    const int n = blockIdx.x;
    const int t = threadIdx.x;
    const float* xr = x + (size_t)n * C_DIM;

    // convert this token's row to bf16 (8 elems / thread)
    {
        float4 a = *(const float4*)(xr + t * 8);
        float4 b = *(const float4*)(xr + t * 8 + 4);
        v8bf o;
        o[0] = (__bf16)a.x; o[1] = (__bf16)a.y; o[2] = (__bf16)a.z; o[3] = (__bf16)a.w;
        o[4] = (__bf16)b.x; o[5] = (__bf16)b.y; o[6] = (__bf16)b.z; o[7] = (__bf16)b.w;
        *(v8bf*)(xb + (size_t)n * C_DIM + t * 8) = o;
    }

    // gate logits: thread t handles expert (t&7), C-slice (t>>3)
    __shared__ float sg[E_NUM][33];
    const int e = t & 7, part = t >> 3;
    float s = 0.f;
    for (int c = part * 64; c < part * 64 + 64; ++c)
        s = fmaf(xr[c], Wg[(size_t)c * E_NUM + e], s);
    sg[e][part] = s;
    __syncthreads();
    if (t < E_NUM) {
        float acc = 0.f;
        for (int p = 0; p < 32; ++p) acc += sg[t][p];
        sg[t][32] = acc;
    }
    __syncthreads();
    if (t == 0) {
        float l[E_NUM];
        #pragma unroll
        for (int i = 0; i < E_NUM; ++i) l[i] = sg[i][32];
        int i0 = 0;
        #pragma unroll
        for (int i = 1; i < E_NUM; ++i) if (l[i] > l[i0]) i0 = i;
        int i1 = (i0 == 0) ? 1 : 0;
        #pragma unroll
        for (int i = 0; i < E_NUM; ++i) if (i != i0 && l[i] > l[i1]) i1 = i;
        // softmax + top-2 renorm collapses to a sigmoid of the logit gap
        float w0 = 1.f / (1.f + __expf(l[i1] - l[i0]));
        float w1 = 1.f - w0;
        int s0 = atomicAdd(&counts[i0], 1);
        tokList[(size_t)i0 * N_TOK + s0] = n;
        wList [(size_t)i0 * N_TOK + s0] = w0;
        int s1 = atomicAdd(&counts[i1], 1);
        tokList[(size_t)i1 * N_TOK + s1] = n;
        wList [(size_t)i1 * N_TOK + s1] = w1;
    }
}

// ---------------------------------------------------------------------------
// Tiled WMMA GEMM: D(cnt x N) = op(A(cnt x K,bf16) * B(K x N,f32->bf16))
//  GATHER : token/weight lists active (cnt from device counter)
//  AGATHER: A rows indexed through token list (expert up-proj reading x)
//  SILU   : apply x*sigmoid(x) to result
//  OUTMODE: 0 = store bf16 (h buffer), 1 = store f32, 2 = weighted atomic f32
template<int GATHER, int AGATHER, int SILU, int OUTMODE>
__global__ __launch_bounds__(256) void moe_gemm_kernel(
    const __bf16* __restrict__ A, int lda,
    const float*  __restrict__ Bw, int ldb, int K,
    void* __restrict__ Dst, int ldd,
    const int* __restrict__ tokList, const float* __restrict__ wList,
    const int* __restrict__ cntPtr, int cntConst)
{
    __shared__ __attribute__((aligned(16))) __bf16 ldsB[BN * LDSS];
    __shared__ int   sTok[BM];
    __shared__ float sW[BM];

    const int cnt = GATHER ? *cntPtr : cntConst;
    const int m0 = blockIdx.y * BM;
    const int n0 = blockIdx.x * BN;
    if (m0 >= cnt) return;   // uniform: early-exit empty tiles

    const int t     = threadIdx.x;
    const int lane  = t & 31;
    const int wave  = t >> 5;          // 8 waves: 4 (M) x 2 (N)
    const int waveM = (wave & 3) * 64;
    const int waveN = (wave >> 2) * 64;

    {
        int gm = m0 + t;
        int tok = gm; float w = 1.f;
        if (GATHER) {
            if (gm < cnt) { tok = tokList[gm]; w = wList[gm]; }
            else          { tok = 0;           w = 0.f; }
        }
        sTok[t] = tok; sW[t] = w;
    }
    __syncthreads();

    // Per-lane A fragment base pointers (WMMA 16x32 bf16 A layout:
    // lane L -> row L%16, 16B chunks at k = (L/16)*8 and +16).
    // Rows >= cnt clamp to row 0: garbage lands only in D rows never stored.
    const __bf16* aPtr[4];
    #pragma unroll
    for (int i = 0; i < 4; ++i) {
        const int ml = waveM + i * 16 + (lane & 15);
        const int gm = m0 + ml;
        int ar;
        if (AGATHER) ar = (gm < cnt) ? sTok[ml] : 0;
        else         ar = (gm < cnt) ? gm : 0;
        aPtr[i] = A + (size_t)ar * lda + ((lane >> 4) << 3);
    }

    v8f acc[4][4];
    #pragma unroll
    for (int i = 0; i < 4; ++i)
        #pragma unroll
        for (int j = 0; j < 4; ++j) acc[i][j] = (v8f)(0.0f);

    const int ffCol = (t & 31) << 2;   // B-stage column group (4 cols)
    const int kPair = t >> 5;          // B-stage k-pair id (0..7)

    for (int k0 = 0; k0 < K; k0 += BK) {
        // ---- stage B tile: 32k x 128n fp32 -> bf16, transposed ldsB[n][k].
        //      Paired k-conversion -> v_cvt_pk_bf16_f32 + ds_store_b32. ----
        #pragma unroll
        for (int r = 0; r < 2; ++r) {
            const int kb = kPair * 2 + r * 16;
            const float* g0 = Bw + (size_t)(k0 + kb)     * ldb + n0 + ffCol;
            const float* g1 = Bw + (size_t)(k0 + kb + 1) * ldb + n0 + ffCol;
            float4 w0 = *(const float4*)g0;
            float4 w1 = *(const float4*)g1;
            v2bf p0; p0[0] = (__bf16)w0.x; p0[1] = (__bf16)w1.x;
            v2bf p1; p1[0] = (__bf16)w0.y; p1[1] = (__bf16)w1.y;
            v2bf p2; p2[0] = (__bf16)w0.z; p2[1] = (__bf16)w1.z;
            v2bf p3; p3[0] = (__bf16)w0.w; p3[1] = (__bf16)w1.w;
            __bf16* d = ldsB + (ffCol)*LDSS + kb;
            *(v2bf*)(d + 0 * LDSS) = p0;
            *(v2bf*)(d + 1 * LDSS) = p1;
            *(v2bf*)(d + 2 * LDSS) = p2;
            *(v2bf*)(d + 3 * LDSS) = p3;
        }
        if (k0 + BK < K)   // prefetch next weight K-chunk -> global_prefetch_b8
            __builtin_prefetch(Bw + (size_t)(k0 + BK + kPair) * ldb + n0 + ffCol, 0, 0);
        __syncthreads();

        // ---- B fragments from LDS (same 16x32 per-lane layout on B^T) ----
        v16bf bfm[4];
        #pragma unroll
        for (int j = 0; j < 4; ++j) {
            const __bf16* p = ldsB + (waveN + j * 16 + (lane & 15)) * LDSS + ((lane >> 4) << 3);
            v8bf lo = *(const v8bf*)p;
            v8bf hi = *(const v8bf*)(p + 16);
            #pragma unroll
            for (int q = 0; q < 8; ++q) { bfm[j][q] = lo[q]; bfm[j][q + 8] = hi[q]; }
        }
        // ---- A fragments straight from global (L2-resident) + 16 WMMAs ----
        #pragma unroll
        for (int i = 0; i < 4; ++i) {
            v8bf lo = *(const v8bf*)(aPtr[i] + k0);
            v8bf hi = *(const v8bf*)(aPtr[i] + k0 + 16);
            v16bf af;
            #pragma unroll
            for (int q = 0; q < 8; ++q) { af[q] = lo[q]; af[q + 8] = hi[q]; }
            #pragma unroll
            for (int j = 0; j < 4; ++j)
                acc[i][j] = __builtin_amdgcn_wmma_f32_16x16x32_bf16(
                    false, af, false, bfm[j], (short)0, acc[i][j], false, false);
        }
        __syncthreads();
    }

    // ---- epilogue: 16x16 f32 C/D layout: lane L -> col L%16, rows (L/16)*8+v ----
    #pragma unroll
    for (int i = 0; i < 4; ++i) {
        #pragma unroll
        for (int j = 0; j < 4; ++j) {
            const int nCol  = n0 + waveN + j * 16 + (lane & 15);
            const int mBase = waveM + i * 16 + ((lane >> 4) << 3);
            #pragma unroll
            for (int v = 0; v < 8; ++v) {
                const int ml = mBase + v;
                const int gm = m0 + ml;
                if (gm < cnt) {
                    float val = acc[i][j][v];
                    if (SILU) val = val / (1.f + __expf(-val));
                    if (OUTMODE == 0) {
                        ((__bf16*)Dst)[(size_t)gm * ldd + nCol] = (__bf16)val;
                    } else if (OUTMODE == 1) {
                        ((float*)Dst)[(size_t)gm * ldd + nCol] = val;
                    } else {
                        atomicAdd(&((float*)Dst)[(size_t)sTok[ml] * ldd + nCol], sW[ml] * val);
                    }
                }
            }
        }
    }
}

// ---------------------------------------------------------------------------
extern "C" void kernel_launch(void* const* d_in, const int* in_sizes, int n_in,
                              void* d_out, int out_size, void* d_ws, size_t ws_size,
                              hipStream_t stream) {
    const float* x   = (const float*)d_in[0];
    const float* Wg  = (const float*)d_in[1];
    const float* W1  = (const float*)d_in[2];
    const float* W2  = (const float*)d_in[3];
    const float* Ws1 = (const float*)d_in[4];
    const float* Ws2 = (const float*)d_in[5];
    float* y = (float*)d_out;

    // workspace layout (~98 MB), fully re-initialized each call
    char* ws = (char*)d_ws;
    size_t off = 0;
    __bf16* xb      = (__bf16*)(ws + off); off += (size_t)N_TOK * C_DIM * sizeof(__bf16);
    __bf16* hb      = (__bf16*)(ws + off); off += (size_t)N_TOK * F_DIM * sizeof(__bf16);
    int*    tokList = (int*)   (ws + off); off += (size_t)E_NUM * N_TOK * sizeof(int);
    float*  wListp  = (float*) (ws + off); off += (size_t)E_NUM * N_TOK * sizeof(float);
    int*    counts  = (int*)   (ws + off);

    zero_counts_kernel<<<1, 32, 0, stream>>>(counts);
    gate_kernel<<<N_TOK, 256, 0, stream>>>(x, Wg, xb, counts, tokList, wListp);

    dim3 blk(256);
    dim3 gUp(F_DIM / BN, N_TOK / BM);   // 32 x 32
    dim3 gDn(C_DIM / BN, N_TOK / BM);   // 16 x 32

    // shared expert: h = silu(x @ Ws1); y = h @ Ws2 (plain stores init y fully)
    moe_gemm_kernel<0,0,1,0><<<gUp, blk, 0, stream>>>(
        xb, C_DIM, Ws1, F_DIM, C_DIM, hb, F_DIM, nullptr, nullptr, nullptr, N_TOK);
    moe_gemm_kernel<0,0,0,1><<<gDn, blk, 0, stream>>>(
        hb, F_DIM, Ws2, C_DIM, F_DIM, y, C_DIM, nullptr, nullptr, nullptr, N_TOK);

    // routed experts (stream-serialized; h buffer reused per expert)
    for (int e = 0; e < E_NUM; ++e) {
        const float* W1e = W1 + (size_t)e * C_DIM * F_DIM;
        const float* W2e = W2 + (size_t)e * F_DIM * C_DIM;
        const int*   tl  = tokList + (size_t)e * N_TOK;
        const float* wl  = wListp  + (size_t)e * N_TOK;
        moe_gemm_kernel<1,1,1,0><<<gUp, blk, 0, stream>>>(
            xb, C_DIM, W1e, F_DIM, C_DIM, hb, F_DIM, tl, wl, counts + e, 0);
        moe_gemm_kernel<1,0,0,2><<<gDn, blk, 0, stream>>>(
            hb, F_DIM, W2e, C_DIM, F_DIM, y, C_DIM, tl, wl, counts + e, 0);
    }
}
